// SANClassifier_40535901340169
// MI455X (gfx1250) — matrix operand order repaired and checked by reading, not compiled, not run on hardware
//
#include <hip/hip_runtime.h>

// Problem constants (match reference)
#define BB   64
#define SP   512
#define SH   512
#define HH   1024
#define CC   3
#define KIT  5
#define H3   (3*HH)

// Workspace layout (floats)
#define OFF_S0  0
#define OFF_X0  (OFF_S0 + BB*HH)
#define OFF_T1  (OFF_X0 + BB*HH)
#define OFF_GX  (OFF_T1 + BB*SP)
#define OFF_GH  (OFF_GX + BB*H3)
#define OFF_PS  (OFF_GH + BB*H3)

typedef __attribute__((ext_vector_type(16))) __bf16 v16bf;
typedef __attribute__((ext_vector_type(8)))  float  v8f;

__device__ __forceinline__ float waveSum(float v) {
    #pragma unroll
    for (int o = 16; o > 0; o >>= 1) v += __shfl_xor(v, o, 32);
    return v;
}
__device__ __forceinline__ float waveMax(float v) {
    #pragma unroll
    for (int o = 16; o > 0; o >>= 1) v = fmaxf(v, __shfl_xor(v, o, 32));
    return v;
}

// ---------------------------------------------------------------------------
// Kernel A: alpha = masked_softmax(hyp . sa_w + sa_b); s0 = alpha . hyp
// one block (1024 threads = 32 waves) per batch element
// ---------------------------------------------------------------------------
__global__ __launch_bounds__(1024)
void kA_selfattn(const float* __restrict__ hyp, const int* __restrict__ h_mask,
                 const float* __restrict__ sa_w, const float* __restrict__ sa_b,
                 float* __restrict__ s0) {
    const int b = blockIdx.x;
    __shared__ float sw[HH];
    __shared__ float sc[SH];
    __shared__ float red[32];
    const int tid = threadIdx.x, lane = tid & 31, wid = tid >> 5;

    sw[tid] = sa_w[tid];
    __syncthreads();

    const float* hb = hyp + (size_t)b * SH * HH;
    const float sab = sa_b[0];

    // scores: one wave per row, 16 rows per wave
    for (int s = wid; s < SH; s += 32) {
        const float* row = hb + (size_t)s * HH;
        float p = 0.f;
        for (int h = lane; h < HH; h += 32) p = fmaf(row[h], sw[h], p);
        p = waveSum(p);
        if (lane == 0) {
            float v = p + sab;
            if (h_mask[b * SH + s] == 1) v = -INFINITY;
            sc[s] = v;
        }
    }
    __syncthreads();

    // softmax over SH
    float v = (tid < SH) ? sc[tid] : -INFINITY;
    float m = waveMax(v);
    if (lane == 0) red[wid] = m;
    __syncthreads();
    if (tid < 32) { float t = red[tid]; t = waveMax(t); if (tid == 0) red[0] = t; }
    __syncthreads();
    m = red[0];
    __syncthreads();
    float e = (tid < SH) ? __expf(v - m) : 0.f;
    float ssum = waveSum(e);
    if (lane == 0) red[wid] = ssum;
    __syncthreads();
    if (tid < 32) { float t = red[tid]; t = waveSum(t); if (tid == 0) red[0] = t; }
    __syncthreads();
    const float inv = 1.f / red[0];
    if (tid < SH) sc[tid] = e * inv;
    __syncthreads();

    // s0[h] = sum_s alpha[s] * hyp[b,s,h]   (column walk, coalesced across tid)
    float acc = 0.f;
    for (int s = 0; s < SH; ++s) acc = fmaf(sc[s], hb[(size_t)s * HH + tid], acc);
    s0[(size_t)b * HH + tid] = acc;
}

// ---------------------------------------------------------------------------
// Kernel B: t1[b,s] = premise[b,s,:] . w1 + at_b   (loop-invariant attn term)
// ---------------------------------------------------------------------------
__global__ __launch_bounds__(1024)
void kB_t1(const float* __restrict__ prem, const float* __restrict__ at_w,
           const float* __restrict__ at_b, float* __restrict__ t1) {
    const int b = blockIdx.x;
    __shared__ float sw[HH];
    const int tid = threadIdx.x, lane = tid & 31, wid = tid >> 5;
    sw[tid] = at_w[tid];                       // w1 = at_w[0:H]
    __syncthreads();
    const float* pb = prem + (size_t)b * SP * HH;
    const float atb = at_b[0];
    for (int s = wid; s < SP; s += 32) {
        const float* row = pb + (size_t)s * HH;
        float p = 0.f;
        for (int h = lane; h < HH; h += 32) p = fmaf(row[h], sw[h], p);
        p = waveSum(p);
        if (lane == 0) t1[b * SP + s] = p + atb;
    }
}

// ---------------------------------------------------------------------------
// Kernel C (per iteration): attn scores + masked softmax + x0 + logits/probs
// ---------------------------------------------------------------------------
__global__ __launch_bounds__(1024)
void kC_attend(const float* __restrict__ prem, const int* __restrict__ p_mask,
               const float* __restrict__ at_w, const float* __restrict__ cls_w,
               const float* __restrict__ cls_b, const float* __restrict__ t1,
               const float* __restrict__ s0, float* __restrict__ x0,
               float* __restrict__ probs_sum) {
    const int b = blockIdx.x;
    __shared__ float s0sh[HH];
    __shared__ float u3[HH];
    __shared__ float w4sh[HH];
    __shared__ float sc[SP];
    __shared__ float red[32];
    __shared__ float lred[32][3];
    __shared__ float svsh;
    const int tid = threadIdx.x, lane = tid & 31, wid = tid >> 5;

    const float s0v = s0[(size_t)b * HH + tid];
    s0sh[tid] = s0v;
    u3[tid]   = s0v * at_w[2 * HH + tid];      // s0*w3
    w4sh[tid] = at_w[3 * HH + tid];            // w4
    // term2: sv = s0 . w2
    float pv = s0v * at_w[HH + tid];
    pv = waveSum(pv);
    if (lane == 0) red[wid] = pv;
    __syncthreads();
    if (tid < 32) { float t = red[tid]; t = waveSum(t); if (tid == 0) svsh = t; }
    __syncthreads();
    const float sv = svsh;

    const float* pb = prem + (size_t)b * SP * HH;
    for (int s = wid; s < SP; s += 32) {
        const float* row = pb + (size_t)s * HH;
        if (s + 32 < SP) __builtin_prefetch(pb + (size_t)(s + 32) * HH + lane * 4, 0, 0);
        float p = 0.f;
        for (int h = lane; h < HH; h += 32) {
            float x = row[h];
            p = fmaf(x, u3[h], p);
            p = fmaf(fabsf(x - s0sh[h]), w4sh[h], p);
        }
        p = waveSum(p);
        if (lane == 0) {
            float vv = t1[b * SP + s] + sv + p;
            if (p_mask[b * SP + s] == 1) vv = -INFINITY;
            sc[s] = vv;
        }
    }
    __syncthreads();

    // masked softmax over SP
    float v = (tid < SP) ? sc[tid] : -INFINITY;
    float m = waveMax(v);
    if (lane == 0) red[wid] = m;
    __syncthreads();
    if (tid < 32) { float t = red[tid]; t = waveMax(t); if (tid == 0) red[0] = t; }
    __syncthreads();
    m = red[0];
    __syncthreads();
    float e = (tid < SP) ? __expf(v - m) : 0.f;
    float ssum = waveSum(e);
    if (lane == 0) red[wid] = ssum;
    __syncthreads();
    if (tid < 32) { float t = red[tid]; t = waveSum(t); if (tid == 0) red[0] = t; }
    __syncthreads();
    const float inv = 1.f / red[0];
    if (tid < SP) sc[tid] = e * inv;
    __syncthreads();

    // x0[h] = sum_s beta[s] * premise[b,s,h]
    float acc = 0.f;
    for (int s = 0; s < SP; ++s) acc = fmaf(sc[s], pb[(size_t)s * HH + tid], acc);
    x0[(size_t)b * HH + tid] = acc;

    // logits[c] = sum_h x0[h]*cls_w[c,h] + s0[h]*cls_w[c,H+h]   (c = 0..2)
    float l0 = acc * cls_w[0 * 2 * HH + tid] + s0v * cls_w[0 * 2 * HH + HH + tid];
    float l1 = acc * cls_w[1 * 2 * HH + tid] + s0v * cls_w[1 * 2 * HH + HH + tid];
    float l2 = acc * cls_w[2 * 2 * HH + tid] + s0v * cls_w[2 * 2 * HH + HH + tid];
    l0 = waveSum(l0); l1 = waveSum(l1); l2 = waveSum(l2);
    if (lane == 0) { lred[wid][0] = l0; lred[wid][1] = l1; lred[wid][2] = l2; }
    __syncthreads();
    if (tid < 32) {
        float a0 = lred[tid][0], a1 = lred[tid][1], a2 = lred[tid][2];
        a0 = waveSum(a0); a1 = waveSum(a1); a2 = waveSum(a2);
        if (tid == 0) {
            a0 += cls_b[0]; a1 += cls_b[1]; a2 += cls_b[2];
            float mm = fmaxf(a0, fmaxf(a1, a2));
            float e0 = __expf(a0 - mm), e1 = __expf(a1 - mm), e2 = __expf(a2 - mm);
            float d = e0 + e1 + e2;
            probs_sum[b * CC + 0] += e0 / d;
            probs_sum[b * CC + 1] += e1 / d;
            probs_sum[b * CC + 2] += e2 / d;
        }
    }
}

// ---------------------------------------------------------------------------
// Kernel D (per iteration): GRU GEMMs via v_wmma_f32_16x16x32_bf16
//   z==0: gx = x0 @ W_ih^T + b_ih   ;  z==1: gh = s0 @ W_hh^T + b_hh
// M=64 (batch), N=3072, K=1024. One wave per 16x16 output tile.
// A 16x32 bf16 layout: lanes 0-15 row M=lane hold K 0..7 / 16..23 (elem 0..7/8..15);
//                      lanes 16-31 same rows with K offset +8.
// B 32x16 bf16 layout: same K order as A (lanes 0-15 K 0..7/16..23; lanes 16-31 +8).
// C/D f32: VGPR r -> M = r (+8 for lanes 16-31), N = lane%16.
// ---------------------------------------------------------------------------
__global__ __launch_bounds__(128)
void kD_gemm(const float* __restrict__ x0, const float* __restrict__ s0,
             const float* __restrict__ W_ih, const float* __restrict__ W_hh,
             const float* __restrict__ b_ih, const float* __restrict__ b_hh,
             float* __restrict__ gx, float* __restrict__ gh) {
    const int lane = threadIdx.x & 31;
    const int wv   = threadIdx.x >> 5;
    const int jbase = (blockIdx.x * 4 + wv) * 16;   // N tile
    const int mbase = blockIdx.y * 16;              // M tile

    const float *A, *W, *bias; float *out;
    if (blockIdx.z == 0) { A = x0; W = W_ih; bias = b_ih; out = gx; }
    else                 { A = s0; W = W_hh; bias = b_hh; out = gh; }

    const int  l16  = lane & 15;
    const int  aoff = (lane >= 16) ? 8 : 0;

    const float* Ap = A + (size_t)(mbase + l16) * HH;
    const float* Wp = W + (size_t)(jbase + l16) * HH;

    v8f c = {};
    for (int kb = 0; kb < HH; kb += 32) {
        v16bf av, bv;
        #pragma unroll
        for (int e = 0; e < 8; ++e) {
            av[e]     = (__bf16)Ap[kb + aoff + e];
            av[e + 8] = (__bf16)Ap[kb + 16 + aoff + e];
        }
        #pragma unroll
        for (int e = 0; e < 8; ++e) {
            bv[e]     = (__bf16)Wp[kb + aoff + e];
            bv[e + 8] = (__bf16)Wp[kb + 16 + aoff + e];
        }
        c = __builtin_amdgcn_wmma_f32_16x16x32_bf16(
                false, av, false, bv, (short)0, c, false, false);
    }

    const int mrow = mbase + ((lane >= 16) ? 8 : 0);
    const int j    = jbase + l16;
    const float bb = bias[j];
    #pragma unroll
    for (int r = 0; r < 8; ++r)
        out[(size_t)(mrow + r) * H3 + j] = c[r] + bb;
}

// ---------------------------------------------------------------------------
// Kernel E (per iteration): GRU gate combine, s0 updated in place
// ---------------------------------------------------------------------------
__global__ __launch_bounds__(256)
void kE_gru(const float* __restrict__ gx, const float* __restrict__ gh,
            float* __restrict__ s0) {
    const int idx = blockIdx.x * blockDim.x + threadIdx.x;   // B*H
    const int b = idx >> 10, h = idx & (HH - 1);
    const float* gxb = gx + (size_t)b * H3;
    const float* ghb = gh + (size_t)b * H3;
    const float xr = gxb[h], xz = gxb[HH + h], xn = gxb[2 * HH + h];
    const float hr = ghb[h], hz = ghb[HH + h], hn = ghb[2 * HH + h];
    const float r = 1.f / (1.f + __expf(-(xr + hr)));
    const float z = 1.f / (1.f + __expf(-(xz + hz)));
    const float n = tanhf(xn + r * hn);
    const float s = s0[idx];
    s0[idx] = (1.f - z) * n + z * s;
}

// ---------------------------------------------------------------------------
// Kernel F: out = log(probs_sum / K)
// ---------------------------------------------------------------------------
__global__ void kF_out(const float* __restrict__ ps, float* __restrict__ out) {
    const int i = blockIdx.x * blockDim.x + threadIdx.x;
    if (i < BB * CC) out[i] = __logf(ps[i] * (1.0f / (float)KIT));
}

// ---------------------------------------------------------------------------
extern "C" void kernel_launch(void* const* d_in, const int* in_sizes, int n_in,
                              void* d_out, int out_size, void* d_ws, size_t ws_size,
                              hipStream_t stream) {
    const float* hyp    = (const float*)d_in[0];
    const float* prem   = (const float*)d_in[1];
    const int*   p_mask = (const int*)  d_in[2];
    const int*   h_mask = (const int*)  d_in[3];
    const float* sa_w   = (const float*)d_in[4];
    const float* sa_b   = (const float*)d_in[5];
    const float* at_w   = (const float*)d_in[6];
    const float* at_b   = (const float*)d_in[7];
    const float* W_ih   = (const float*)d_in[8];
    const float* W_hh   = (const float*)d_in[9];
    const float* b_ih   = (const float*)d_in[10];
    const float* b_hh   = (const float*)d_in[11];
    const float* cls_w  = (const float*)d_in[12];
    const float* cls_b  = (const float*)d_in[13];

    float* ws = (float*)d_ws;
    float* s0 = ws + OFF_S0;
    float* x0 = ws + OFF_X0;
    float* t1 = ws + OFF_T1;
    float* gx = ws + OFF_GX;
    float* gh = ws + OFF_GH;
    float* ps = ws + OFF_PS;

    hipMemsetAsync(ps, 0, BB * CC * sizeof(float), stream);

    kA_selfattn<<<BB, 1024, 0, stream>>>(hyp, h_mask, sa_w, sa_b, s0);
    kB_t1<<<BB, 1024, 0, stream>>>(prem, at_w, at_b, t1);

    for (int it = 0; it < KIT; ++it) {
        kC_attend<<<BB, 1024, 0, stream>>>(prem, p_mask, at_w, cls_w, cls_b,
                                           t1, s0, x0, ps);
        kD_gemm<<<dim3(48, 4, 2), 128, 0, stream>>>(x0, s0, W_ih, W_hh,
                                                    b_ih, b_hh, gx, gh);
        kE_gru<<<(BB * HH) / 256, 256, 0, stream>>>(gx, gh, s0);
    }

    kF_out<<<1, 256, 0, stream>>>(ps, (float*)d_out);
}
